// RNNClassifier_11905649345044
// MI455X (gfx1250) — matrix operand order repaired and checked by reading, not compiled, not run on hardware
//
#include <hip/hip_runtime.h>

// ---------------------------------------------------------------------------
// GRU twin-tower classifier for gfx1250 (MI455X).
//   S=512, B=512, E=50, H=128.  Recurrence is batch-parallel: 16-row batch
//   tiles per workgroup, weights resident in LDS, f16 WMMA / f32 accumulate.
//   Embedding gather is double-buffered and prefetched one step ahead so the
//   two chained L2 round trips (index -> row) hide under the WMMA phase.
// ---------------------------------------------------------------------------

typedef __attribute__((ext_vector_type(16))) _Float16 v16h;
typedef __attribute__((ext_vector_type(8)))  _Float16 v8h;
typedef __attribute__((ext_vector_type(8)))  float    v8f;

#define S_LEN 512
#define BATCH 512
#define EMB_D 50
#define EPAD  64      // E padded to 2 K-chunks of 32
#define HDIM  128
#define G3    384     // 3*H
#define BT    16      // batch tile (one WMMA M tile)

// Build a 16-element f16 A/B fragment from two contiguous 16-byte LDS chunks.
__device__ __forceinline__ v16h ld_frag(const _Float16* p0, const _Float16* p1) {
  v8h lo = *(const v8h*)p0;
  v8h hi = *(const v8h*)p1;
  v16h r;
#pragma unroll
  for (int i = 0; i < 8; ++i) { r[i] = lo[i]; r[i + 8] = hi[i]; }
  return r;
}

__device__ __forceinline__ v8f wmma_f16(v16h a, v16h b, v8f c) {
  // D = A(16x32 f16) * B(32x16 f16) + C(16x16 f32)
  return __builtin_amdgcn_wmma_f32_16x16x32_f16(false, a, false, b,
                                                (short)0, c, false, false);
}

__device__ __forceinline__ float sigf(float x) {
  return __builtin_amdgcn_rcpf(1.f + __expf(-x));
}
__device__ __forceinline__ float tanhf_fast(float x) {
  float e2 = __expf(2.f * x);
  return 1.f - 2.f * __builtin_amdgcn_rcpf(e2 + 1.f);
}

// grid = (BATCH/BT, 2)  block = 256 (8 waves)
__global__ __launch_bounds__(256, 1)
void gru_kernel(const int* __restrict__ x1, const int* __restrict__ x2,
                const float* __restrict__ emb,
                const float* __restrict__ W_ih, const float* __restrict__ W_hh,
                const float* __restrict__ b_ih, const float* __restrict__ b_hh,
                float* __restrict__ h_out /* (2, BATCH, H) */) {
  // LDS-resident weights + state (~155.6 KB -> one workgroup per WGP)
  __shared__ __align__(16) _Float16 sWhh[G3 * HDIM];   // [g][k]  96 KB
  __shared__ __align__(16) _Float16 sWih[G3 * EPAD];   // [g][e]  48 KB
  __shared__ __align__(16) _Float16 sH[BT * HDIM];     // h tile f16  4 KB
  __shared__ __align__(16) _Float16 sExA[BT * EPAD];   // emb ping  2 KB
  __shared__ __align__(16) _Float16 sExB[BT * EPAD];   // emb pong  2 KB

  const int tid  = threadIdx.x;
  const int lane = tid & 31;
  const int wv   = tid >> 5;        // wave 0..7 -> owns cols [wv*16, wv*16+16)
  const int ln   = lane & 15;       // N (and A row) index within tile
  const int hi   = lane >> 4;       // K-half selector
  const int bbase = blockIdx.x * BT;
  const int* __restrict__ x = (blockIdx.y == 0) ? x1 : x2;

  // per-thread gather assignment: 4 consecutive e slots in one row m
  const int gm = tid >> 4;          // row 0..15
  const int ge = (tid & 15) * 4;    // e base 0,4,...,60

  // ---- stage weights (f32 -> f16) -----------------------------------------
  for (int i = tid; i < G3 * HDIM; i += 256) sWhh[i] = (_Float16)W_hh[i];
  for (int i = tid; i < G3 * EPAD; i += 256) {
    int g = i >> 6, e = i & 63;
    sWih[i] = (e < EMB_D) ? (_Float16)W_ih[g * EMB_D + e] : (_Float16)0.f;
  }
  for (int i = tid; i < BT * HDIM; i += 256) sH[i] = (_Float16)0.f;

  // ---- gather step 0 embeddings into ping buffer --------------------------
  {
    const int idx0 = x[bbase + gm];
#pragma unroll
    for (int k = 0; k < 4; ++k) {
      float v = (ge + k < EMB_D) ? emb[idx0 * EMB_D + ge + k] : 0.f;
      sExA[gm * EPAD + ge + k] = (_Float16)v;
    }
  }

  // per-lane hidden column j and its biases (stable across all steps)
  const int   j   = wv * 16 + ln;            // 0..127
  const float bR  = b_ih[j]          + b_hh[j];
  const float bZ  = b_ih[HDIM + j]   + b_hh[HDIM + j];
  const float bIN = b_ih[2 * HDIM + j];
  const float bHN = b_hh[2 * HDIM + j];
  v8f hreg = {};                             // h rows m = v + 8*hi, col j (f32)
  __syncthreads();                           // weights + step-0 sEx visible

  for (int s = 0; s < S_LEN; ++s) {
    const bool odd = s & 1;
    const _Float16* cur = odd ? sExB : sExA;
    _Float16*       nxt = odd ? sExA : sExB;

    // ---- issue next step's gather early (hidden under WMMA phase) ---------
    const int sp = (s + 1 < S_LEN) ? s + 1 : s;     // clamp on last iter
    const int pidx = x[sp * BATCH + bbase + gm];
    float pf[4];
#pragma unroll
    for (int k = 0; k < 4; ++k)
      pf[k] = (ge + k < EMB_D) ? emb[pidx * EMB_D + ge + k] : 0.f;

    // ---- A fragments: h (4 K-chunks of 32), ex (2 K-chunks) ---------------
    v16h ah[4], ae[2];
#pragma unroll
    for (int c = 0; c < 4; ++c) {
      const _Float16* p = &sH[ln * HDIM + c * 32 + hi * 16];
      ah[c] = ld_frag(p, p + 8);
    }
#pragma unroll
    for (int c = 0; c < 2; ++c) {
      const _Float16* p = &cur[ln * EPAD + c * 32 + hi * 16];
      ae[c] = ld_frag(p, p + 8);
    }

    // ---- 18 WMMAs: r/z (combined x+h proj), n (x and h kept separate) -----
    v8f accR = {}, accZ = {}, accNX = {}, accNH = {};
#pragma unroll
    for (int c = 0; c < 4; ++c) {            // h @ Whh^T, K = 128
      const int k0 = c * 32;
      const _Float16* pr = &sWhh[(wv * 16 + ln) * HDIM + k0 + hi * 16];
      accR  = wmma_f16(ah[c], ld_frag(pr, pr + 8), accR);
      const _Float16* pz = &sWhh[(HDIM + wv * 16 + ln) * HDIM + k0 + hi * 16];
      accZ  = wmma_f16(ah[c], ld_frag(pz, pz + 8), accZ);
      const _Float16* pn = &sWhh[(2 * HDIM + wv * 16 + ln) * HDIM + k0 + hi * 16];
      accNH = wmma_f16(ah[c], ld_frag(pn, pn + 8), accNH);
    }
#pragma unroll
    for (int c = 0; c < 2; ++c) {            // ex @ Wih^T, K = 64 (padded)
      const int k0 = c * 32;
      const _Float16* pr = &sWih[(wv * 16 + ln) * EPAD + k0 + hi * 16];
      accR  = wmma_f16(ae[c], ld_frag(pr, pr + 8), accR);
      const _Float16* pz = &sWih[(HDIM + wv * 16 + ln) * EPAD + k0 + hi * 16];
      accZ  = wmma_f16(ae[c], ld_frag(pz, pz + 8), accZ);
      const _Float16* pn = &sWih[(2 * HDIM + wv * 16 + ln) * EPAD + k0 + hi * 16];
      accNX = wmma_f16(ae[c], ld_frag(pn, pn + 8), accNX);
    }
    __syncthreads();   // all waves done reading sH / cur before overwrite

    // ---- gate math in registers; r/z/n of unit j share this lane ----------
#pragma unroll
    for (int v = 0; v < 8; ++v) {
      float r = sigf(accR[v] + bR);
      float z = sigf(accZ[v] + bZ);
      float n = tanhf_fast(accNX[v] + bIN + r * (accNH[v] + bHN));
      hreg[v] = (1.f - z) * n + z * hreg[v];
      sH[(v + 8 * hi) * HDIM + j] = (_Float16)hreg[v];   // f16 copy for next A
    }

    // ---- commit prefetched embeddings for step s+1 ------------------------
#pragma unroll
    for (int k = 0; k < 4; ++k)
      nxt[gm * EPAD + ge + k] = (_Float16)pf[k];

    __syncthreads();   // sH + next sEx visible for step s+1
  }

  // ---- write final hidden state (f32) -------------------------------------
  float* ho = h_out + (size_t)blockIdx.y * BATCH * HDIM;
#pragma unroll
  for (int v = 0; v < 8; ++v)
    ho[(bbase + v + 8 * hi) * HDIM + j] = hreg[v];
}

// grid = BATCH, block = 128: out = sigmoid(relu([h1,h2]@W1^T + b1)@W2^T + b2)
__global__ __launch_bounds__(128)
void mlp_kernel(const float* __restrict__ h12,
                const float* __restrict__ W1, const float* __restrict__ b1,
                const float* __restrict__ W2, const float* __restrict__ b2,
                float* __restrict__ out) {
  __shared__ float red[128];
  const int b = blockIdx.x, t = threadIdx.x;
  const float* h1 = h12 + b * HDIM;
  const float* h2 = h12 + BATCH * HDIM + b * HDIM;
  float acc = b1[t];
  const float* w = W1 + t * 2 * HDIM;
#pragma unroll 4
  for (int k = 0; k < HDIM; ++k) acc += h1[k] * w[k];
#pragma unroll 4
  for (int k = 0; k < HDIM; ++k) acc += h2[k] * w[HDIM + k];
  red[t] = fmaxf(acc, 0.f) * W2[t];
  __syncthreads();
  for (int off = 64; off > 0; off >>= 1) {
    if (t < off) red[t] += red[t + off];
    __syncthreads();
  }
  if (t == 0) out[b] = sigf(red[0] + b2[0]);
}

extern "C" void kernel_launch(void* const* d_in, const int* in_sizes, int n_in,
                              void* d_out, int out_size, void* d_ws, size_t ws_size,
                              hipStream_t stream) {
  const int*   x1   = (const int*)  d_in[0];
  const int*   x2   = (const int*)  d_in[1];
  const float* emb  = (const float*)d_in[2];
  const float* W_ih = (const float*)d_in[3];
  const float* W_hh = (const float*)d_in[4];
  const float* b_ih = (const float*)d_in[5];
  const float* b_hh = (const float*)d_in[6];
  const float* W1   = (const float*)d_in[7];
  const float* b1   = (const float*)d_in[8];
  const float* W2   = (const float*)d_in[9];
  const float* b2   = (const float*)d_in[10];
  float* out  = (float*)d_out;
  float* hbuf = (float*)d_ws;   // 2 * BATCH * HDIM f32 = 512 KB

  gru_kernel<<<dim3(BATCH / BT, 2), 256, 0, stream>>>(
      x1, x2, emb, W_ih, W_hh, b_ih, b_hh, hbuf);
  mlp_kernel<<<dim3(BATCH), 128, 0, stream>>>(hbuf, W1, b1, W2, b2, out);
}
